// self_attention_80891414053563
// MI455X (gfx1250) — matrix operand order repaired and checked by reading, not compiled, not run on hardware
//
#include <hip/hip_runtime.h>
#include <hip/hip_bf16.h>

// ---------------- problem constants ----------------
constexpr int kB     = 16;
constexpr int kS     = 2048;
constexpr int kD     = 512;
constexpr int kHeads = 8;
constexpr int kDK    = 64;

// ---------------- types ----------------
typedef __bf16          bf16x16 __attribute__((ext_vector_type(16)));
typedef float           f32x8   __attribute__((ext_vector_type(8)));
typedef unsigned short  u16x8   __attribute__((ext_vector_type(8)));
typedef int             v4i     __attribute__((ext_vector_type(4)));

#define AS1 __attribute__((address_space(1)))
#define AS3 __attribute__((address_space(3)))

union Frag {             // 32 bytes = 8 VGPRs = one WMMA bf16 A/B fragment
    bf16x16 v;
    u16x8   h[2];
    uint4   q[2];
};

// ---------------- CDNA5 feature probes ----------------
#if __has_builtin(__builtin_amdgcn_global_load_async_to_lds_b128) && \
    __has_builtin(__builtin_amdgcn_s_wait_asynccnt)
#define USE_ASYNC_LDS 1
#else
#define USE_ASYNC_LDS 0
#endif

#if __has_builtin(__builtin_amdgcn_perm)
#define USE_PERM 1
#else
#define USE_PERM 0
#endif

// fp32 -> bf16 bits, round-to-nearest (ties away): 1 VALU + shift
__device__ __forceinline__ unsigned short bfbits(float f) {
    unsigned u = __builtin_bit_cast(unsigned, f);
    return (unsigned short)((u + 0x8000u) >> 16);
}

// two fp32 -> packed bf16x2: 2 adds + 1 v_perm_b32
__device__ __forceinline__ unsigned cvt2bits(float lo, float hi) {
    unsigned ulo = __builtin_bit_cast(unsigned, lo) + 0x8000u;
    unsigned uhi = __builtin_bit_cast(unsigned, hi) + 0x8000u;
#if USE_PERM
    // result bytes: [1:0] = ulo[3:2], [3:2] = uhi[3:2]  (a=uhi -> bytes 4..7)
    return __builtin_amdgcn_perm(uhi, ulo, 0x07060302u);
#else
    return (ulo >> 16) | (uhi & 0xFFFF0000u);
#endif
}

__device__ __forceinline__ u16x8 cvt8(const float* p) {
    float4 f0 = *(const float4*)p;
    float4 f1 = *(const float4*)(p + 4);
    union { u16x8 h; unsigned u[4]; } r;
    r.u[0] = cvt2bits(f0.x, f0.y);
    r.u[1] = cvt2bits(f0.z, f0.w);
    r.u[2] = cvt2bits(f1.x, f1.y);
    r.u[3] = cvt2bits(f1.z, f1.w);
    return r.h;
}

__device__ __forceinline__ f32x8 wmma_bf16(bf16x16 a, bf16x16 b, f32x8 c) {
    return __builtin_amdgcn_wmma_f32_16x16x32_bf16(
        false, a, false, b, (short)0, c, false, false);
}

// 16-byte global -> LDS copy: async (ASYNCcnt) when available, else sync
__device__ __forceinline__ void cp16_g2l(const unsigned short* g, unsigned short* l) {
#if USE_ASYNC_LDS
    __builtin_amdgcn_global_load_async_to_lds_b128(
        (AS1 v4i*)(uintptr_t)g,
        (AS3 v4i*)(unsigned)(uintptr_t)l,
        0, 0);
#else
    *(uint4*)l = *(const uint4*)g;
#endif
}

__device__ __forceinline__ void async_drain() {
#if USE_ASYNC_LDS
    __builtin_amdgcn_s_wait_asynccnt(0);
#endif
}

// ================================================================
// Kernel 0: W[d][e] fp32  ->  Wt[e][d] bf16  (x3 weights)
// ================================================================
__global__ __launch_bounds__(256) void wconv_kernel(
    const float* __restrict__ Wq, const float* __restrict__ Wk,
    const float* __restrict__ Wv,
    unsigned short* __restrict__ WtQ, unsigned short* __restrict__ WtK,
    unsigned short* __restrict__ WtV) {
    int idx = blockIdx.x * 256 + threadIdx.x;
    if (idx >= kD * kD) return;
    int e = idx >> 9, d = idx & (kD - 1);
    int src = d * kD + e;
    WtQ[idx] = bfbits(Wq[src]);
    WtK[idx] = bfbits(Wk[src]);
    WtV[idx] = bfbits(Wv[src]);
}

// ================================================================
// Kernel 1: fused QKV projection, software-pipelined.
//   Per wave: one 16-row M tile x 32-col N group, 6 accumulators.
//   Double-buffered fragments: iteration i+1's A + 6 B fragments are
//   loaded before iteration i's WMMAs so loads overlap math.
// ================================================================
__global__ __launch_bounds__(256) void qkv_kernel(
    const float* __restrict__ x,
    const unsigned short* __restrict__ WtQ,
    const unsigned short* __restrict__ WtK,
    const unsigned short* __restrict__ WtV,
    const float* __restrict__ bq, const float* __restrict__ bk,
    const float* __restrict__ bv,
    unsigned short* __restrict__ Qb, unsigned short* __restrict__ Kb,
    unsigned short* __restrict__ Vt) {
    const int lane = threadIdx.x & 31;
    const int half = lane >> 4;
    const int ln   = lane & 15;
    const int wid  = blockIdx.x * 8 + (threadIdx.x >> 5);
    const int mt   = wid >> 4;          // 16-row tile index over B*S
    const int n0   = (wid & 15) * 32;   // 32-col output group

    const f32x8 z = {};
    f32x8 aq[2] = {z, z}, ak[2] = {z, z}, av[2] = {z, z};

    const long   arow = (long)(mt * 16 + ln);
    const float* xrow = x + arow * kD;
    const int    woff0 = (n0 + ln) * kD + half * 16;        // sub 0 B row base
    const int    woff1 = (n0 + 16 + ln) * kD + half * 16;   // sub 1 B row base

    Frag a[2], wq[2][2], wk[2][2], wv[2][2];

    auto load_stage = [&](int it, int pb) {
        const int kk = it * 32;
        const float* xp = xrow + kk + half * 16;
        a[pb].h[0] = cvt8(xp);
        a[pb].h[1] = cvt8(xp + 8);
        const int o0 = woff0 + kk, o1 = woff1 + kk;
        wq[pb][0].q[0] = *(const uint4*)(WtQ + o0); wq[pb][0].q[1] = *(const uint4*)(WtQ + o0 + 8);
        wq[pb][1].q[0] = *(const uint4*)(WtQ + o1); wq[pb][1].q[1] = *(const uint4*)(WtQ + o1 + 8);
        wk[pb][0].q[0] = *(const uint4*)(WtK + o0); wk[pb][0].q[1] = *(const uint4*)(WtK + o0 + 8);
        wk[pb][1].q[0] = *(const uint4*)(WtK + o1); wk[pb][1].q[1] = *(const uint4*)(WtK + o1 + 8);
        wv[pb][0].q[0] = *(const uint4*)(WtV + o0); wv[pb][0].q[1] = *(const uint4*)(WtV + o0 + 8);
        wv[pb][1].q[0] = *(const uint4*)(WtV + o1); wv[pb][1].q[1] = *(const uint4*)(WtV + o1 + 8);
    };

    constexpr int NK = kD / 32;         // 16 K-chunks
    load_stage(0, 0);
    for (int it = 0; it < NK; ++it) {
        const int pb = it & 1;
        if (it + 1 < NK) load_stage(it + 1, pb ^ 1);
#pragma unroll
        for (int sub = 0; sub < 2; ++sub) {
            aq[sub] = wmma_bf16(a[pb].v, wq[pb][sub].v, aq[sub]);
            ak[sub] = wmma_bf16(a[pb].v, wk[pb][sub].v, ak[sub]);
            av[sub] = wmma_bf16(a[pb].v, wv[pb][sub].v, av[sub]);
        }
    }

#pragma unroll
    for (int sub = 0; sub < 2; ++sub) {
        const int   col = n0 + sub * 16 + ln;     // C/D layout: lane -> column
        const float biq = bq[col], bik = bk[col], biv = bv[col];
        const int   hh = col >> 6, ei = col & (kDK - 1);
#pragma unroll
        for (int r = 0; r < 8; ++r) {             // VGPR r -> row r (+8 hi half)
            const long row = (long)(mt * 16) + r + half * 8;
            const int  bb  = (int)(row >> 11), s = (int)(row & (kS - 1));
            Qb[row * kD + col] = bfbits((aq[sub][r] + biq) * 0.125f);
            Kb[row * kD + col] = bfbits(ak[sub][r] + bik);
            Vt[(((long)bb * kHeads + hh) * kDK + ei) * kS + s] =
                bfbits(av[sub][r] + biv);
        }
    }
}

// ================================================================
// Kernel 2: flash attention with async double-buffered LDS staging.
//   Block = 8 waves = 128 queries for one (b, h).
//   K/V 32-key tiles prefetched into the alternate LDS buffer via
//   global_load_async_to_lds_b128 while the current tile computes;
//   s_wait_asynccnt(0) only at the iteration-end barrier.
// ================================================================
__global__ __launch_bounds__(256) void attn_kernel(
    const unsigned short* __restrict__ Qb,
    const unsigned short* __restrict__ Kb,
    const unsigned short* __restrict__ Vt,
    float* __restrict__ out) {
    __shared__ unsigned short kt [2][32 * kDK];   // [buf][key][e]   8 KB
    __shared__ unsigned short vts[2][kDK * 32];   // [buf][e][key]   8 KB
    __shared__ unsigned short ps [8][16 * 32];    // per-wave P      8 KB

    const int tid  = threadIdx.x;
    const int wave = tid >> 5;
    const int lane = tid & 31;
    const int half = lane >> 4;
    const int ln   = lane & 15;

    const int blk = blockIdx.x;
    const int qb  = blk & 15;
    const int h   = (blk >> 4) & 7;
    const int b   = blk >> 7;

    const int  q0   = qb * 128 + wave * 16;
    const long qrow = (long)(b * kS + q0 + ln);

    // Q A-fragments for e-chunks 0..31 and 32..63 (pre-scaled by 1/8)
    Frag qf[2];
#pragma unroll
    for (int c = 0; c < 2; ++c) {
        const unsigned short* qp = Qb + qrow * kD + h * kDK + c * 32;
        qf[c].q[0] = *(const uint4*)(qp + half * 16);
        qf[c].q[1] = *(const uint4*)(qp + half * 16 + 8);
    }

    float m[8], l[8];
    f32x8 o[4];
    const f32x8 z = {};
#pragma unroll
    for (int r = 0; r < 8; ++r) { m[r] = -3.0e38f; l[r] = 0.0f; }
#pragma unroll
    for (int t = 0; t < 4; ++t) o[t] = z;

    const long kbase = (long)b * kS;                     // K rows
    const long vbase = (long)(b * kHeads + h) * kDK;     // Vt rows (len S)

    const int kkey = tid >> 3, kec = tid & 7;   // K-tile stage coords
    const int ve   = tid >> 2, vkc = tid & 3;   // V-tile stage coords

    auto stage = [&](int key0, int buf) {
        cp16_g2l(Kb + (kbase + key0 + kkey) * kD + h * kDK + kec * 8,
                 &kt[buf][kkey * kDK + kec * 8]);
        cp16_g2l(Vt + (vbase + ve) * kS + key0 + vkc * 8,
                 &vts[buf][ve * 32 + vkc * 8]);
    };

    stage(0, 0);
    async_drain();
    __syncthreads();

    int buf = 0;
    for (int key0 = 0; key0 < kS; key0 += 32) {
        // prefetch next tile into the other buffer (overlaps all math below)
        if (key0 + 32 < kS) stage(key0 + 32, buf ^ 1);

        // ---- sim tile: load all 4 B-fragments, then 4 WMMAs ----
        Frag kb[2][2];
#pragma unroll
        for (int c = 0; c < 2; ++c) {
            const int kk = c * 32;
#pragma unroll
            for (int t = 0; t < 2; ++t) {
                const int off = (t * 16 + ln) * kDK + kk + half * 16;
                kb[c][t].q[0] = *(const uint4*)&kt[buf][off];
                kb[c][t].q[1] = *(const uint4*)&kt[buf][off + 8];
            }
        }
        f32x8 s0 = z, s1 = z;
        s0 = wmma_bf16(qf[0].v, kb[0][0].v, s0);
        s1 = wmma_bf16(qf[0].v, kb[0][1].v, s1);
        s0 = wmma_bf16(qf[1].v, kb[1][0].v, s0);
        s1 = wmma_bf16(qf[1].v, kb[1][1].v, s1);

        // ---- online softmax (rows live in 16-lane halves) ----
        unsigned short* pw = ps[wave];
#pragma unroll
        for (int r = 0; r < 8; ++r) {
            float a0 = s0[r], a1 = s1[r];
            float mx = fmaxf(a0, a1);
#pragma unroll
            for (int msk = 1; msk < 16; msk <<= 1) mx = fmaxf(mx, __shfl_xor(mx, msk));
            float mn    = fmaxf(m[r], mx);
            float alpha = __expf(m[r] - mn);
            m[r] = mn;
            float p0 = __expf(a0 - mn);
            float p1 = __expf(a1 - mn);
            float rs = p0 + p1;
#pragma unroll
            for (int msk = 1; msk < 16; msk <<= 1) rs += __shfl_xor(rs, msk);
            l[r] = l[r] * alpha + rs;
#pragma unroll
            for (int t = 0; t < 4; ++t) o[t][r] *= alpha;
            int row = r + half * 8;
            // store packed pair is impossible (keys land in different lanes):
            // two b16 stores with cheap rounding instead
            pw[row * 32 + ln]      = bfbits(p0);
            pw[row * 32 + 16 + ln] = bfbits(p1);
        }

        // ---- O += P * V: load P A-frag + 4 V B-frags, then 4 WMMAs ----
        Frag pa, vb[4];
        pa.q[0] = *(const uint4*)&pw[ln * 32 + half * 16];
        pa.q[1] = *(const uint4*)&pw[ln * 32 + half * 16 + 8];
#pragma unroll
        for (int t = 0; t < 4; ++t) {
            const int off = (t * 16 + ln) * 32 + half * 16;
            vb[t].q[0] = *(const uint4*)&vts[buf][off];
            vb[t].q[1] = *(const uint4*)&vts[buf][off + 8];
        }
#pragma unroll
        for (int t = 0; t < 4; ++t) o[t] = wmma_bf16(pa.v, vb[t].v, o[t]);

        async_drain();      // prefetched tile fully in LDS
        __syncthreads();    // all waves done with current buffers
        buf ^= 1;
    }

    // ---- finalize: divide by l, store fp32 ----
#pragma unroll
    for (int r = 0; r < 8; ++r) {
        float inv = 1.0f / l[r];
        long  row = (long)(b * kS + q0 + r + half * 8);
#pragma unroll
        for (int t = 0; t < 4; ++t) {
            int col = h * kDK + t * 16 + ln;
            out[row * kD + col] = o[t][r] * inv;
        }
    }
}

// ================================================================
extern "C" void kernel_launch(void* const* d_in, const int* in_sizes, int n_in,
                              void* d_out, int out_size, void* d_ws, size_t ws_size,
                              hipStream_t stream) {
    const float* x  = (const float*)d_in[0];
    const float* Wq = (const float*)d_in[1];
    const float* bq = (const float*)d_in[2];
    const float* Wk = (const float*)d_in[3];
    const float* bk = (const float*)d_in[4];
    const float* Wv = (const float*)d_in[5];
    const float* bv = (const float*)d_in[6];
    float* out = (float*)d_out;

    // workspace layout (bf16): 3 transposed weights + Q + K + V^T  (~97.5 MB)
    unsigned short* WtQ = (unsigned short*)d_ws;
    unsigned short* WtK = WtQ + (size_t)kD * kD;
    unsigned short* WtV = WtK + (size_t)kD * kD;
    unsigned short* Qb  = WtV + (size_t)kD * kD;
    unsigned short* Kb  = Qb + (size_t)kB * kS * kD;
    unsigned short* Vt  = Kb + (size_t)kB * kS * kD;

    wconv_kernel<<<(kD * kD + 255) / 256, 256, 0, stream>>>(Wq, Wk, Wv, WtQ, WtK, WtV);

    qkv_kernel<<<(kB * kS / 16) * (kD / 32) / 8, 256, 0, stream>>>(
        x, WtQ, WtK, WtV, bq, bk, bv, Qb, Kb, Vt);

    attn_kernel<<<kB * kHeads * (kS / 128), 256, 0, stream>>>(Qb, Kb, Vt, out);
}
